// SuperpixelCoreModel_16681652978287
// MI455X (gfx1250) — matrix operand order repaired and checked by reading, not compiled, not run on hardware
//
#include <hip/hip_runtime.h>

// ---- problem constants (from reference) ----
static constexpr int NB  = 8;       // batch
static constexpr int NC  = 384;     // channels
static constexpr int HWp = 50176;   // 224*224
static constexpr int NS  = 256;     // superpixel labels
static constexpr int NM  = 65536;   // memory bank rows
static constexpr int PB  = 8;       // pixel blocks per batch (segment partials)
static constexpr int CCH = 32;      // channels per segment-reduce chunk
static constexpr int NROW = NB * NS; // 2048 emb rows

typedef __attribute__((ext_vector_type(16))) __bf16 v16bf;
typedef __attribute__((ext_vector_type(8)))  __bf16 v8bf;
typedef __attribute__((ext_vector_type(8)))  float  v8f;

__device__ __forceinline__ unsigned long long umin64(unsigned long long a, unsigned long long b) { return a < b ? a : b; }
__device__ __forceinline__ unsigned long long umax64(unsigned long long a, unsigned long long b) { return a > b ? a : b; }

__device__ __forceinline__ unsigned long long shfl_xor_u64(unsigned long long v, int m) {
    unsigned lo = (unsigned)v, hi = (unsigned)(v >> 32);
    lo = __shfl_xor(lo, m, 32);
    hi = __shfl_xor(hi, m, 32);
    return ((unsigned long long)hi << 32) | lo;
}

// A fragment (16x32 bf16, MxK): lane L (L<16): row M=L, K {0..7} then {16..23};
// lanes 16-31: same rows, K {8..15} then {24..31}.  p points at (row, k0 + half*8).
__device__ __forceinline__ v16bf load_frag_a(const __bf16* p) {
    v8bf lo = *(const v8bf*)(p);
    v8bf hi = *(const v8bf*)(p + 16);
    v16bf r;
#pragma unroll
    for (int i = 0; i < 8; ++i) { r[i] = lo[i]; r[i + 8] = hi[i]; }
    return r;
}

// B fragment (32x16 bf16, KxN): lane L holds column N=L&15; lanes 0-15: K {0..7} then {16..23},
// lanes 16-31: K {8..15} then {24..31}.  p points at (col, k0 + half*8).
__device__ __forceinline__ v16bf load_frag_b(const __bf16* p) {
    v8bf lo = *(const v8bf*)(p);
    v8bf hi = *(const v8bf*)(p + 16);
    v16bf r;
#pragma unroll
    for (int i = 0; i < 8; ++i) { r[i] = lo[i]; r[i + 8] = hi[i]; }
    return r;
}

// ---------------------------------------------------------------------------
// 0) init best keys to +inf
__global__ void k_init_best(unsigned long long* __restrict__ best) {
    best[blockIdx.x * 256 + threadIdx.x] = ~0ULL;
}

// ---------------------------------------------------------------------------
// 1) segment partial sums + counts (LDS accumulation, deterministic partials)
__global__ void k_segpart(const float* __restrict__ feats, const int* __restrict__ labels,
                          float* __restrict__ part, int* __restrict__ cntp) {
    __shared__ float acc[CCH * 257];   // padded: c*257 + lab -> conflict-free readback
    __shared__ int   labs[256];
    __shared__ int   cnt[256];
    const int pb = blockIdx.x, cz = blockIdx.y, b = blockIdx.z;
    const int tid = threadIdx.x;
    const int c0 = cz * CCH;
    for (int i = tid; i < CCH * 257; i += 256) acc[i] = 0.0f;
    if (cz == 0) cnt[tid] = 0;
    __syncthreads();

    const int pcount = HWp / PB;            // 6272
    const int pstart = pb * pcount;
    const float* fbase = feats + (size_t)b * NC * HWp;
    const int*   lbase = labels + (size_t)b * HWp;

    for (int t0 = 0; t0 < pcount; t0 += 256) {
        const int pofs = t0 + tid;
        const bool valid = pofs < pcount;
        const int p = pstart + pofs;
        labs[tid] = valid ? (lbase[p] & 255) : 0;
        __syncthreads();
        if (valid) {
            const int lab = labs[tid];
#pragma unroll 8
            for (int c = 0; c < CCH; ++c) {
                float v = fbase[(size_t)(c0 + c) * HWp + p];
                atomicAdd(&acc[c * 257 + lab], v);      // ds_add_f32
            }
            if (cz == 0) atomicAdd(&cnt[lab], 1);
        }
        __syncthreads();
    }

    float* pdst = part + (((size_t)pb * NB + b) * NS) * NC;
    for (int i = tid; i < NS * CCH; i += 256) {
        const int lab = i / CCH, c = i % CCH;
        pdst[(size_t)lab * NC + c0 + c] = acc[c * 257 + lab];
    }
    if (cz == 0) cntp[((size_t)pb * NB + b) * NS + tid] = cnt[tid];
}

// ---------------------------------------------------------------------------
// 2) finalize emb = seg_sum / max(count,1); emit fp32, bf16, and row norms
__global__ void k_embfin(const float* __restrict__ part, const int* __restrict__ cntp,
                         float* __restrict__ embf, __bf16* __restrict__ embbf,
                         float* __restrict__ embnorm) {
    const int row = blockIdx.x;            // b*NS + s
    const int b = row >> 8, s = row & 255;
    const int tid = threadIdx.x;
    __shared__ float red[128];
    __shared__ float cinv;
    if (tid == 0) {
        int c = 0;
        for (int pb = 0; pb < PB; ++pb) c += cntp[((size_t)pb * NB + b) * NS + s];
        cinv = 1.0f / (float)(c > 1 ? c : 1);
    }
    __syncthreads();
    float nrm = 0.0f;
    for (int c = tid; c < NC; c += 128) {
        float v = 0.0f;
        for (int pb = 0; pb < PB; ++pb)
            v += part[(((size_t)pb * NB + b) * NS + s) * NC + c];
        v *= cinv;
        embf[(size_t)row * NC + c]  = v;
        embbf[(size_t)row * NC + c] = (__bf16)v;
        nrm += v * v;
    }
    red[tid] = nrm; __syncthreads();
    for (int w = 64; w > 0; w >>= 1) { if (tid < w) red[tid] += red[tid + w]; __syncthreads(); }
    if (tid == 0) embnorm[row] = red[0];
}

// ---------------------------------------------------------------------------
// 3) memory bank: bf16 copy + row norms
__global__ void k_bankprep(const float* __restrict__ bank, __bf16* __restrict__ bankbf,
                           float* __restrict__ banknorm) {
    const int row = blockIdx.x;
    const int tid = threadIdx.x;
    __shared__ float red[128];
    float nrm = 0.0f;
    for (int c = tid; c < NC; c += 128) {
        float v = bank[(size_t)row * NC + c];
        bankbf[(size_t)row * NC + c] = (__bf16)v;
        nrm += v * v;
    }
    red[tid] = nrm; __syncthreads();
    for (int w = 64; w > 0; w >>= 1) { if (tid < w) red[tid] += red[tid + w]; __syncthreads(); }
    if (tid == 0) banknorm[row] = red[0];
}

// ---------------------------------------------------------------------------
// 4) WMMA distance kernel: fused min/argmin over 2048x65536 distances.
//    Block: 256 thr (8 waves). Block tile: 32 emb rows x 512 bank cols.
//    Wave tile: 2 row-tiles x 4 col-tiles of 16x16, K=384 (12 x wmma bf16 k32).
__global__ void __launch_bounds__(256, 2)
k_dist(const __bf16* __restrict__ embbf, const __bf16* __restrict__ bankbf,
       const float* __restrict__ embnorm, const float* __restrict__ banknorm,
       unsigned long long* __restrict__ best) {
    __shared__ __bf16 ldsA[32 * NC];                       // 24 KB
    const int tid  = threadIdx.x;
    const int wave = tid >> 5;
    const int lane = tid & 31;
    const int half = lane >> 4;        // 0: lanes 0-15, 1: lanes 16-31
    const int lmod = lane & 15;
    const int row0 = blockIdx.y * 32;
    const int col0 = blockIdx.x * 512 + wave * 64;

    { // cooperative A-tile load (32 rows x 384 bf16, contiguous)
        const unsigned* src = (const unsigned*)(embbf + (size_t)row0 * NC);
        unsigned* dst = (unsigned*)ldsA;
        for (int i = tid; i < 32 * NC / 2; i += 256) dst[i] = src[i];
    }
    __syncthreads();

    v8f acc[2][4];
#pragma unroll
    for (int rt = 0; rt < 2; ++rt)
#pragma unroll
        for (int ct = 0; ct < 4; ++ct)
#pragma unroll
            for (int i = 0; i < 8; ++i) acc[rt][ct][i] = 0.0f;

#pragma unroll 1
    for (int kk = 0; kk < NC / 32; ++kk) {
        const int akofs = kk * 32 + half * 8;
        const int bkofs = kk * 32 + half * 8;
        v16bf a0 = load_frag_a(&ldsA[(0  + lmod) * NC + akofs]);
        v16bf a1 = load_frag_a(&ldsA[(16 + lmod) * NC + akofs]);
#pragma unroll
        for (int ct = 0; ct < 4; ++ct) {
            const int ncol = col0 + ct * 16 + lmod;
            v16bf bf = load_frag_b(&bankbf[(size_t)ncol * NC + bkofs]);
            acc[0][ct] = __builtin_amdgcn_wmma_f32_16x16x32_bf16(
                false, a0, false, bf, (short)0, acc[0][ct], false, false);
            acc[1][ct] = __builtin_amdgcn_wmma_f32_16x16x32_bf16(
                false, a1, false, bf, (short)0, acc[1][ct], false, false);
        }
    }

    // epilogue: d = sqrt(max(0, xn + yn - 2*dot)); key = (bits(d)<<32)|col; min-merge
#pragma unroll
    for (int rt = 0; rt < 2; ++rt) {
#pragma unroll
        for (int r = 0; r < 8; ++r) {
            const int m = row0 + rt * 16 + r + half * 8;    // C/D layout: VGPR r -> M=r (+8 for hi half)
            const float xn = embnorm[m];
            unsigned long long kmin = ~0ULL;
#pragma unroll
            for (int ct = 0; ct < 4; ++ct) {
                const int n = col0 + ct * 16 + lmod;        // C/D layout: N = lane&15
                float d2 = xn + banknorm[n] - 2.0f * acc[rt][ct][r];
                d2 = fmaxf(d2, 0.0f);
                const float d = __builtin_sqrtf(d2);
                const unsigned long long key =
                    ((unsigned long long)__float_as_uint(d) << 32) | (unsigned)n;
                kmin = umin64(kmin, key);
            }
#pragma unroll
            for (int sh = 1; sh < 16; sh <<= 1)             // reduce within 16-lane half
                kmin = umin64(kmin, shfl_xor_u64(kmin, sh));
            if (lmod == 0) atomicMin(&best[m], kmin);
        }
    }
}

// ---------------------------------------------------------------------------
// 5) per-batch argmax of scores (tie -> lowest s), gather nn index
__global__ void k_argmax(const unsigned long long* __restrict__ best,
                         int* __restrict__ maxrow, float* __restrict__ score,
                         int* __restrict__ nnidx) {
    const int b = blockIdx.x, tid = threadIdx.x;           // tid == s
    __shared__ unsigned long long red[256];
    const unsigned long long key = best[b * NS + tid];
    const unsigned dbits = (unsigned)(key >> 32);
    red[tid] = ((unsigned long long)dbits << 32) | (unsigned)(NS - 1 - tid);
    __syncthreads();
    for (int w = 128; w > 0; w >>= 1) { if (tid < w) red[tid] = umax64(red[tid], red[tid + w]); __syncthreads(); }
    if (tid == 0) {
        const unsigned long long r = red[0];
        const int s = NS - 1 - (int)(r & 0xFFFFFFFFu);
        const int row = b * NS + s;
        maxrow[b] = row;
        score[b]  = __uint_as_float((unsigned)(r >> 32));
        nnidx[b]  = (int)(unsigned)(best[row] & 0xFFFFFFFFu);
    }
}

// ---------------------------------------------------------------------------
// 6) d_nn[b][m] = ||bank[nn_b] - bank[m]||; one bank pass serves all 8 batches
__global__ void k_dnn(const float* __restrict__ bank, const int* __restrict__ nnidx,
                      float* __restrict__ dnn) {
    __shared__ float nnf[NB * NC];                         // 12 KB
    __shared__ int   nns[NB];
    const int tid = threadIdx.x;
    if (tid < NB) nns[tid] = nnidx[tid];
    __syncthreads();
    for (int i = tid; i < NB * NC; i += 256) {
        const int j = i / NC, c = i % NC;
        nnf[i] = bank[(size_t)nns[j] * NC + c];
    }
    __syncthreads();
    const int wave = tid >> 5, lane = tid & 31;
    const int m = blockIdx.x * 8 + wave;
    float a[NB] = {0, 0, 0, 0, 0, 0, 0, 0};
    for (int c = lane; c < NC; c += 32) {
        const float v = bank[(size_t)m * NC + c];
#pragma unroll
        for (int j = 0; j < NB; ++j) { const float df = v - nnf[j * NC + c]; a[j] += df * df; }
    }
#pragma unroll
    for (int j = 0; j < NB; ++j) {
        float s = a[j];
        for (int sh = 1; sh < 32; sh <<= 1) s += __shfl_xor(s, sh, 32);
        if (lane == 0) dnn[(size_t)j * NM + m] = __builtin_sqrtf(s);
    }
}

// ---------------------------------------------------------------------------
// 7) per-batch: exact top-9 smallest (value, index) of d_nn, softmax reweight
__global__ void k_stage2(const float* __restrict__ dnn, const float* __restrict__ bank,
                         const float* __restrict__ embf, const int* __restrict__ maxrow,
                         const float* __restrict__ score, float* __restrict__ outpred) {
    const int b = blockIdx.x, tid = threadIdx.x;
    __shared__ unsigned long long redk[256];
    __shared__ float redf[256];
    __shared__ int   support[9];
    __shared__ float dsup[9];
    __shared__ unsigned long long prev;

    for (int k = 0; k < 9; ++k) {
        unsigned long long lm = ~0ULL;
        const unsigned long long pk = (k == 0) ? 0ULL : prev;
        for (int m = tid; m < NM; m += 256) {
            const float d = dnn[(size_t)b * NM + m];
            const unsigned long long key =
                ((unsigned long long)__float_as_uint(d) << 32) | (unsigned)m;
            if (k == 0 || key > pk) lm = umin64(lm, key);
        }
        redk[tid] = lm; __syncthreads();
        for (int w = 128; w > 0; w >>= 1) { if (tid < w) redk[tid] = umin64(redk[tid], redk[tid + w]); __syncthreads(); }
        if (tid == 0) { prev = redk[0]; support[k] = (int)(unsigned)(redk[0] & 0xFFFFFFFFu); }
        __syncthreads();
    }

    const int row = maxrow[b];
    for (int k = 0; k < 9; ++k) {
        float a = 0.0f;
        for (int c = tid; c < NC; c += 256) {
            const float df = embf[(size_t)row * NC + c] - bank[(size_t)support[k] * NC + c];
            a += df * df;
        }
        redf[tid] = a; __syncthreads();
        for (int w = 128; w > 0; w >>= 1) { if (tid < w) redf[tid] += redf[tid + w]; __syncthreads(); }
        if (tid == 0) dsup[k] = __builtin_sqrtf(redf[0]);
        __syncthreads();
    }

    if (tid == 0) {
        float mx = dsup[0];
        for (int k = 1; k < 9; ++k) mx = fmaxf(mx, dsup[k]);
        float sum = 0.0f;
        for (int k = 0; k < 9; ++k) sum += __expf(dsup[k] - mx);
        const float w0 = __expf(dsup[0] - mx) / sum;
        outpred[b] = (1.0f - w0) * score[b];
    }
}

// ---------------------------------------------------------------------------
// 8) scatter superpixel scores to pixel anomaly map
__global__ void k_map(const unsigned long long* __restrict__ best,
                      const int* __restrict__ labels, float* __restrict__ outmap) {
    __shared__ float sc[NS];
    const int b = blockIdx.y, tid = threadIdx.x;
    sc[tid] = __uint_as_float((unsigned)(best[b * NS + tid] >> 32));
    __syncthreads();
    const int p = blockIdx.x * 256 + tid;
    outmap[(size_t)b * HWp + p] = sc[labels[(size_t)b * HWp + p] & 255];
}

// ---------------------------------------------------------------------------
extern "C" void kernel_launch(void* const* d_in, const int* in_sizes, int n_in,
                              void* d_out, int out_size, void* d_ws, size_t ws_size,
                              hipStream_t stream) {
    (void)in_sizes; (void)n_in; (void)out_size; (void)ws_size;
    const float* features   = (const float*)d_in[0];   // (8,384,224,224)
    const int*   superpixel = (const int*)d_in[1];     // (8,224,224)
    const float* membank    = (const float*)d_in[2];   // (65536,384)
    float* out_map  = (float*)d_out;                   // (8,224,224)
    float* out_pred = (float*)d_out + (size_t)NB * HWp;

    char* ws = (char*)d_ws;
    auto take = [&](size_t bytes) { char* p = ws; ws += (bytes + 255) & ~(size_t)255; return p; };
    __bf16* bankbf   = (__bf16*)take((size_t)NM * NC * 2);      // 50.3 MB
    __bf16* embbf    = (__bf16*)take((size_t)NROW * NC * 2);    //  1.6 MB
    float*  embf     = (float*) take((size_t)NROW * NC * 4);    //  3.1 MB
    float*  banknorm = (float*) take((size_t)NM * 4);           //  0.26 MB
    float*  embnorm  = (float*) take((size_t)NROW * 4);
    float*  part     = (float*) take((size_t)PB * NB * NS * NC * 4); // 25.2 MB
    int*    cntp     = (int*)   take((size_t)PB * NB * NS * 4);
    unsigned long long* best = (unsigned long long*)take((size_t)NROW * 8);
    float*  dnn      = (float*) take((size_t)NB * NM * 4);      //  2.1 MB
    int*    maxrow   = (int*)   take(256);
    float*  score    = (float*) take(256);
    int*    nnidx    = (int*)   take(256);

    k_init_best<<<dim3(NROW / 256), dim3(256), 0, stream>>>(best);
    k_segpart  <<<dim3(PB, NC / CCH, NB), dim3(256), 0, stream>>>(features, superpixel, part, cntp);
    k_embfin   <<<dim3(NROW), dim3(128), 0, stream>>>(part, cntp, embf, embbf, embnorm);
    k_bankprep <<<dim3(NM), dim3(128), 0, stream>>>(membank, bankbf, banknorm);
    k_dist     <<<dim3(NM / 512, NROW / 32), dim3(256), 0, stream>>>(embbf, bankbf, embnorm, banknorm, best);
    k_argmax   <<<dim3(NB), dim3(256), 0, stream>>>(best, maxrow, score, nnidx);
    k_dnn      <<<dim3(NM / 8), dim3(256), 0, stream>>>(membank, nnidx, dnn);
    k_stage2   <<<dim3(NB), dim3(256), 0, stream>>>(dnn, membank, embf, maxrow, score, out_pred);
    k_map      <<<dim3(HWp / 256, NB), dim3(256), 0, stream>>>(best, superpixel, out_map);
}
